// MultiHeadedAttention_35510789604074
// MI455X (gfx1250) — matrix operand order repaired and checked by reading, not compiled, not run on hardware
//
#include <hip/hip_runtime.h>
#include <math.h>

// ---------------------------------------------------------------------------
// CDNA5 (gfx1250) fused multi-head "diagonal attention"
//   q = x @ Wq^T ; k = x @ Wk^T ; v = x @ Wv^T        (bf16x3 split WMMA GEMM)
//   diag[b,h,n] = softmax_row_n(q k^T / 8)[n]         (online softmax, WMMA)
//   out0 = diag * v ; out1 = x
// ---------------------------------------------------------------------------

typedef __attribute__((ext_vector_type(16))) __bf16 v16bf;
typedef __attribute__((ext_vector_type(8)))  float  v8f;

union FragB16 {
    v16bf v;
    uint4 u[2];
};

#define WMMA_BF16(A, B, C) \
    __builtin_amdgcn_wmma_f32_16x16x32_bf16(false, (A), false, (B), (short)0, (C), false, false)

__device__ __forceinline__ unsigned bf16_rne(float f) {
    unsigned u = __float_as_uint(f);
    return (u + 0x7FFFu + ((u >> 16) & 1u)) >> 16;
}

// ---------------------------------------------------------------------------
// Split fp32 -> bf16 hi + bf16 lo (residual), round-to-nearest-even.
// ---------------------------------------------------------------------------
__global__ __launch_bounds__(256) void split_bf16_kernel(
    const float* __restrict__ src,
    unsigned short* __restrict__ hi,
    unsigned short* __restrict__ lo,
    int n)
{
    int i = blockIdx.x * blockDim.x + threadIdx.x;
    if (i >= n) return;
    float f = src[i];
    unsigned h = bf16_rne(f);
    float fh = __uint_as_float(h << 16);
    unsigned l = bf16_rne(f - fh);
    hi[i] = (unsigned short)h;
    lo[i] = (unsigned short)l;
}

// ---------------------------------------------------------------------------
// C[m,n] = sum_k A[m,k]*B[n,k]   (A:[M,K] row-major, B:[N,K] row-major)
// bf16x3: A ~ Ah+Al, B ~ Bh+Bl;  C += Ah*Bh + Ah*Bl + Al*Bh
// Block: 256 threads = 8 wave32, block tile 128x128, wave tile 32x64.
// mode 0: store fp32 C.   mode 1: store split bf16 hi/lo (for q,k).
// ---------------------------------------------------------------------------
__global__ __launch_bounds__(256) void gemm_bf16x3_kernel(
    const unsigned short* __restrict__ Ah, const unsigned short* __restrict__ Al,
    const unsigned short* __restrict__ Bh, const unsigned short* __restrict__ Bl,
    float* __restrict__ Cf,
    unsigned short* __restrict__ Ch, unsigned short* __restrict__ Cl,
    int M, int N, int K, int mode)
{
    const int lane  = threadIdx.x & 31;
    const int wave  = threadIdx.x >> 5;
    const int waveM = wave & 3;          // 4 waves along M: 4*32 = 128
    const int waveN = wave >> 2;         // 2 waves along N: 2*64 = 128
    const int half  = lane >> 4;
    const int l16   = lane & 15;

    const int mBase = blockIdx.x * 128 + waveM * 32;
    const int nBase = blockIdx.y * 128 + waveN * 64;

    v8f acc[2][4];
#pragma unroll
    for (int a = 0; a < 2; ++a)
#pragma unroll
        for (int b = 0; b < 4; ++b)
            acc[a][b] = (v8f){0.f, 0.f, 0.f, 0.f, 0.f, 0.f, 0.f, 0.f};

    for (int kb = 0; kb < K; kb += 32) {
        // A fragments (16-bit A layout: K runs [kb+8*half,+8) and [kb+16+8*half,+8))
        FragB16 ahf[2], alf[2];
#pragma unroll
        for (int tm = 0; tm < 2; ++tm) {
            const size_t row = (size_t)(mBase + tm * 16 + l16) * (size_t)K;
            const unsigned short* pa = Ah + row + kb + 8 * half;
            const unsigned short* pl = Al + row + kb + 8 * half;
            ahf[tm].u[0] = *(const uint4*)(pa);
            ahf[tm].u[1] = *(const uint4*)(pa + 16);
            alf[tm].u[0] = *(const uint4*)(pl);
            alf[tm].u[1] = *(const uint4*)(pl + 16);
        }
#pragma unroll
        for (int tn = 0; tn < 4; ++tn) {
            const size_t row = (size_t)(nBase + tn * 16 + l16) * (size_t)K;
            // B layout: K runs [kb+8*half,+8) and [kb+16+8*half,+8) (same as A)
            const unsigned short* pbh = Bh + row + kb + 8 * half;
            const unsigned short* pbl = Bl + row + kb + 8 * half;
            FragB16 bhf, blf;
            bhf.u[0] = *(const uint4*)(pbh);
            bhf.u[1] = *(const uint4*)(pbh + 16);
            blf.u[0] = *(const uint4*)(pbl);
            blf.u[1] = *(const uint4*)(pbl + 16);
#pragma unroll
            for (int tm = 0; tm < 2; ++tm) {
                acc[tm][tn] = WMMA_BF16(ahf[tm].v, bhf.v, acc[tm][tn]);
                acc[tm][tn] = WMMA_BF16(ahf[tm].v, blf.v, acc[tm][tn]);
                acc[tm][tn] = WMMA_BF16(alf[tm].v, bhf.v, acc[tm][tn]);
            }
        }
    }

    // Epilogue. C layout: VGPR j -> row (j + 8*half), col l16 within tile.
#pragma unroll
    for (int tm = 0; tm < 2; ++tm)
#pragma unroll
        for (int tn = 0; tn < 4; ++tn)
#pragma unroll
            for (int j = 0; j < 8; ++j) {
                const int m = mBase + tm * 16 + j + 8 * half;
                const int n = nBase + tn * 16 + l16;
                const size_t idx = (size_t)m * (size_t)N + n;
                const float c = acc[tm][tn][j];
                if (mode == 0) {
                    Cf[idx] = c;
                } else {
                    unsigned h = bf16_rne(c);
                    float fh = __uint_as_float(h << 16);
                    Ch[idx] = (unsigned short)h;
                    Cl[idx] = (unsigned short)bf16_rne(c - fh);
                }
            }
}

// ---------------------------------------------------------------------------
// Per (b,h): S = Q K^T / 8 over 2048x2048 scores, reduced on the fly to
// row (max, sumexp) + diagonal element. diag[b,h,n] = exp(s_nn-max)/sum.
// One 16-row tile per wave; 128 column tiles; 6 WMMAs (bf16x3 x K=64) each.
// ---------------------------------------------------------------------------
__global__ __launch_bounds__(256) void diag_softmax_kernel(
    const unsigned short* __restrict__ qh, const unsigned short* __restrict__ ql,
    const unsigned short* __restrict__ kh, const unsigned short* __restrict__ kl,
    float* __restrict__ diag, int N /*2048*/)
{
    const int bh   = blockIdx.y;           // b*16 + h
    const int b    = bh >> 4;
    const int h    = bh & 15;
    const int lane = threadIdx.x & 31;
    const int wave = threadIdx.x >> 5;
    const int half = lane >> 4;
    const int l16  = lane & 15;

    const int rBase = (blockIdx.x * 8 + wave) * 16;   // 16 rows per wave
    const size_t E  = 1024;
    const size_t headBase = (size_t)b * (size_t)N * E + (size_t)h * 64;

    // Q fragments: persistent across the column loop (hi/lo x 2 K-chunks)
    FragB16 qhf[2], qlf[2];
    {
        const unsigned short* pq = qh + headBase + (size_t)(rBase + l16) * E;
        const unsigned short* pl = ql + headBase + (size_t)(rBase + l16) * E;
#pragma unroll
        for (int c = 0; c < 2; ++c) {
            qhf[c].u[0] = *(const uint4*)(pq + 32 * c + 8 * half);
            qhf[c].u[1] = *(const uint4*)(pq + 32 * c + 16 + 8 * half);
            qlf[c].u[0] = *(const uint4*)(pl + 32 * c + 8 * half);
            qlf[c].u[1] = *(const uint4*)(pl + 32 * c + 16 + 8 * half);
        }
    }

    float mx[8], sm[8], dg[8];
#pragma unroll
    for (int j = 0; j < 8; ++j) { mx[j] = -INFINITY; sm[j] = 0.f; dg[j] = -INFINITY; }

    for (int ct = 0; ct < N / 16; ++ct) {
        const int col = ct * 16 + l16;
        const unsigned short* pk = kh + headBase + (size_t)col * E;
        const unsigned short* pl = kl + headBase + (size_t)col * E;
        FragB16 khf[2], klf[2];
#pragma unroll
        for (int c = 0; c < 2; ++c) {
            khf[c].u[0] = *(const uint4*)(pk + 32 * c + 8 * half);
            khf[c].u[1] = *(const uint4*)(pk + 32 * c + 16 + 8 * half);
            klf[c].u[0] = *(const uint4*)(pl + 32 * c + 8 * half);
            klf[c].u[1] = *(const uint4*)(pl + 32 * c + 16 + 8 * half);
        }
        v8f s = (v8f){0.f, 0.f, 0.f, 0.f, 0.f, 0.f, 0.f, 0.f};
#pragma unroll
        for (int c = 0; c < 2; ++c) {
            s = WMMA_BF16(qhf[c].v, khf[c].v, s);
            s = WMMA_BF16(qhf[c].v, klf[c].v, s);
            s = WMMA_BF16(qlf[c].v, khf[c].v, s);
        }
        // Online softmax update; lane holds col (ct*16+l16) of rows rBase+j+8*half
#pragma unroll
        for (int j = 0; j < 8; ++j) {
            const float sv = s[j] * 0.125f;             // / sqrt(64)
            const int r = rBase + j + 8 * half;
            if (col == r) dg[j] = sv;                   // diagonal capture
            const float nm = fmaxf(mx[j], sv);
            sm[j] = sm[j] * __expf(mx[j] - nm) + __expf(sv - nm);
            mx[j] = nm;
        }
    }

    // Segmented reduction across the 16 lanes sharing each row.
#pragma unroll
    for (int off = 1; off < 16; off <<= 1) {
#pragma unroll
        for (int j = 0; j < 8; ++j) {
            const float om = __shfl_xor(mx[j], off, 32);
            const float os = __shfl_xor(sm[j], off, 32);
            const float od = __shfl_xor(dg[j], off, 32);
            const float nm = fmaxf(mx[j], om);
            sm[j] = sm[j] * __expf(mx[j] - nm) + os * __expf(om - nm);
            mx[j] = nm;
            dg[j] = fmaxf(dg[j], od);
        }
    }
    if (l16 == 0) {
#pragma unroll
        for (int j = 0; j < 8; ++j) {
            const int r = rBase + j + 8 * half;
            diag[(size_t)bh * (size_t)N + r] = __expf(dg[j] - mx[j]) / sm[j];
        }
    }
}

// ---------------------------------------------------------------------------
// out[i] (currently v) *= diag[b,h,n];  out[total+i] = x[i]
// i = ((b*2048 + n)*1024 + h*64 + d)
// ---------------------------------------------------------------------------
__global__ __launch_bounds__(256) void finalize_kernel(
    const float* __restrict__ x, const float* __restrict__ diag,
    float* __restrict__ out, int total)
{
    int i = blockIdx.x * blockDim.x + threadIdx.x;
    if (i >= total) return;
    const int e  = i & 1023;
    const int bn = i >> 10;
    const int n  = bn & 2047;
    const int b  = bn >> 11;
    const int hh = e >> 6;
    const float d = diag[((size_t)(b * 16 + hh)) * 2048 + n];
    out[i] *= d;
    out[(size_t)total + i] = x[i];
}

// ---------------------------------------------------------------------------
extern "C" void kernel_launch(void* const* d_in, const int* in_sizes, int n_in,
                              void* d_out, int out_size, void* d_ws, size_t ws_size,
                              hipStream_t stream) {
    (void)in_sizes; (void)n_in; (void)out_size; (void)ws_size;
    const float* x  = (const float*)d_in[0];
    const float* Wq = (const float*)d_in[1];
    const float* Wk = (const float*)d_in[2];
    const float* Wv = (const float*)d_in[3];
    float* out = (float*)d_out;

    const int Bb = 4, Nn = 2048, E = 1024;
    const int M = Bb * Nn;            // 8192
    const size_t XE = (size_t)M * E;  // 8,388,608
    const size_t WE = (size_t)E * E;  // 1,048,576

    char* ws = (char*)d_ws;
    size_t off = 0;
    unsigned short* xh  = (unsigned short*)(ws + off); off += XE * 2;
    unsigned short* xl  = (unsigned short*)(ws + off); off += XE * 2;
    unsigned short* wh[3], *wl[3];
    for (int i = 0; i < 3; ++i) { wh[i] = (unsigned short*)(ws + off); off += WE * 2; }
    for (int i = 0; i < 3; ++i) { wl[i] = (unsigned short*)(ws + off); off += WE * 2; }
    unsigned short* qh = (unsigned short*)(ws + off); off += XE * 2;
    unsigned short* ql = (unsigned short*)(ws + off); off += XE * 2;
    unsigned short* kh = (unsigned short*)(ws + off); off += XE * 2;
    unsigned short* kl = (unsigned short*)(ws + off); off += XE * 2;
    float* diag = (float*)(ws + off);  // 64 * 2048 floats

    // 1) split x, Wq, Wk, Wv into bf16 hi/lo planes
    split_bf16_kernel<<<(int)((XE + 255) / 256), 256, 0, stream>>>(x, xh, xl, (int)XE);
    split_bf16_kernel<<<(int)((WE + 255) / 256), 256, 0, stream>>>(Wq, wh[0], wl[0], (int)WE);
    split_bf16_kernel<<<(int)((WE + 255) / 256), 256, 0, stream>>>(Wk, wh[1], wl[1], (int)WE);
    split_bf16_kernel<<<(int)((WE + 255) / 256), 256, 0, stream>>>(Wv, wh[2], wl[2], (int)WE);

    // 2) QKV projection GEMMs (bf16x3 WMMA). q,k -> split bf16; v -> out[0:XE] fp32
    dim3 gg(M / 128, E / 128);
    gemm_bf16x3_kernel<<<gg, 256, 0, stream>>>(xh, xl, wh[0], wl[0], nullptr, qh, ql, M, E, E, 1);
    gemm_bf16x3_kernel<<<gg, 256, 0, stream>>>(xh, xl, wh[1], wl[1], nullptr, kh, kl, M, E, E, 1);
    gemm_bf16x3_kernel<<<gg, 256, 0, stream>>>(xh, xl, wh[2], wl[2], out, nullptr, nullptr, M, E, E, 0);

    // 3) fused score GEMM + online-softmax diagonal
    diag_softmax_kernel<<<dim3(Nn / 128, Bb * 16), 256, 0, stream>>>(qh, ql, kh, kl, diag, Nn);

    // 4) out0 = diag * v (in place), out1 = x
    finalize_kernel<<<(int)((XE + 255) / 256), 256, 0, stream>>>(x, diag, out, (int)XE);
}
